// EncoderSRNN_60301340835884
// MI455X (gfx1250) — compile-verified, matrix operand order checked by reading, not run on hardware
//
#include <hip/hip_runtime.h>
#include <hip/hip_bf16.h>
#include <math.h>

// ---------------- problem constants ----------------
#define T_      256
#define B_      128
#define H_      2048
#define NSTACK_ 4
#define S_      64
#define E_      256
#define DEPTH_  4

// GEMM geometry: X (B x K) @ W (K x N)
//  K layout:  [0,2048) emb | [2048,4096) hidden | [4096,8192) stack_vals (n*1024 + d*256 + e)
//  N layout:  [0,2048) mid | [2048,3072) push (n*256+e) | [3072,3104) act (n*3+a, padded to 32)
#define K_      8192
#define KT_     256       // K_/32 k-tiles
#define NTOT_   3104
#define NT_     194       // NTOT_/16 n-tiles (W blocks)
#define NT2_    97        // NTOT_/32 (GEMM workgroups, 2 n-tiles each)
#define MT_     8         // B_/16 m-tiles
#define KB_     4         // k-tiles staged per barrier round

typedef __attribute__((ext_vector_type(16))) __bf16 v16bf;
typedef __attribute__((ext_vector_type(8)))  float  v8f;

// round-to-nearest-even f32 -> bf16
__device__ __forceinline__ __bf16 f2bf(float f) {
    unsigned int u = __float_as_uint(f);
    unsigned int r = (u + 0x7FFFu + ((u >> 16) & 1u)) >> 16;
    unsigned short s = (unsigned short)r;
    __bf16 o;
    __builtin_memcpy(&o, &s, 2);
    return o;
}

// ISA 7.12.2: 16-bit A/B fragment, 16x32 (or 32x16) block stored as 32 lanes x 16 elems.
// lane<16 : j 0..7 -> k 0..7,  j 8..15 -> k 16..23
// lane>=16: j 0..7 -> k 8..15, j 8..15 -> k 24..31
__device__ __forceinline__ int frag_k(int lane, int j) {
    return ((j >> 3) << 4) + ((lane >= 16) ? 8 : 0) + (j & 7);
}

// ------------- once-per-call: swizzled bf16 mega-weight -------------
// block (nt,kt) -> wfrag[(nt*KT_+kt)*512 + lane*16 + j]
__global__ __launch_bounds__(256) void build_wfrag(
    __bf16* __restrict__ wfrag,
    const float* __restrict__ i2h_W, const float* __restrict__ h2h_W,
    const float* __restrict__ s2h_W, const float* __restrict__ h2s_W,
    const float* __restrict__ h2a_W)
{
    int bid = blockIdx.x;              // nt*KT_ + kt
    int nt  = bid / KT_;
    int kt  = bid % KT_;
    #pragma unroll
    for (int half = 0; half < 2; ++half) {
        int p    = threadIdx.x + half * 256;        // 0..511
        int lane = p >> 4;
        int j    = p & 15;
        int k    = kt * 32 + frag_k(lane, j);
        int col  = nt * 16 + (lane & 15);
        float v  = 0.0f;
        if (col < H_) {                              // mid columns, full K
            if (k < 2048)       v = i2h_W[(size_t)k * H_ + col];
            else if (k < 4096)  v = h2h_W[(size_t)(k - 2048) * H_ + col];
            else {
                int kk = k - 4096, n = kk >> 10, d = kk & 1023;
                v = s2h_W[((size_t)(n << 10) + d) * H_ + col];
            }
        } else if (col < 3072) {                     // push columns, hidden K only
            if (k >= 2048 && k < 4096) {
                int cc = col - 2048, n = cc >> 8, ee = cc & 255;
                v = h2s_W[((size_t)n * H_ + (k - 2048)) * E_ + ee];
            }
        } else {                                     // act columns (12 used, rest pad)
            int cc = col - 3072;
            if (cc < 12 && k >= 2048 && k < 4096) {
                int n = cc / 3, a = cc % 3;
                v = h2a_W[((size_t)n * H_ + (k - 2048)) * 3 + a];
            }
        }
        wfrag[(size_t)bid * 512 + p] = f2bf(v);
    }
}

__global__ void build_bias(
    float* __restrict__ bias_mid, float* __restrict__ bias_pn, float* __restrict__ bias_act,
    const float* __restrict__ i2h_b, const float* __restrict__ h2h_b,
    const float* __restrict__ s2h_b, const float* __restrict__ h2s_b,
    const float* __restrict__ h2a_b)
{
    int id = blockIdx.x * blockDim.x + threadIdx.x;
    if (id < H_) {
        float s = 0.f;
        #pragma unroll
        for (int n = 0; n < NSTACK_; ++n) s += s2h_b[n * H_ + id];
        bias_mid[id] = i2h_b[id] + h2h_b[id] + s;
    } else if (id < H_ + 1024) {
        bias_pn[id - H_] = h2s_b[id - H_];           // (4,256) flat
    } else if (id < H_ + 1024 + 32) {
        int c = id - H_ - 1024;
        bias_act[c] = (c < 12) ? h2a_b[c] : 0.f;     // (4,3) flat + pad
    }
}

// ------------- per-step: build bf16 A fragments of X (B x K) -------------
__global__ __launch_bounds__(256) void build_xfrag(
    __bf16* __restrict__ xfrag,
    const int*   __restrict__ inputs,
    const float* __restrict__ embedding,
    const float* __restrict__ hprev,
    const float* __restrict__ stacks,
    int t)
{
    int bid = blockIdx.x;              // mt*KT_ + kt
    int mt  = bid / KT_;
    int kt  = bid % KT_;
    #pragma unroll
    for (int half = 0; half < 2; ++half) {
        int p    = threadIdx.x + half * 256;
        int lane = p >> 4;
        int j    = p & 15;
        int k    = kt * 32 + frag_k(lane, j);
        int b    = mt * 16 + (lane & 15);
        float v;
        if (k < 2048) {
            int tok = inputs[t * B_ + b];
            v = embedding[(size_t)tok * H_ + k];
        } else if (k < 4096) {
            v = hprev[(size_t)b * H_ + (k - 2048)];
        } else {
            int kk = k - 4096;
            int n = kk >> 10, d = (kk >> 8) & 3, ee = kk & 255;
            v = stacks[(((size_t)(b * NSTACK_ + n) * S_ + d) << 8) + ee];
        }
        xfrag[(size_t)bid * 512 + p] = f2bf(v);
    }
}

// ------------- per-step fused GEMM: mid | push | act, WMMA bf16 -------------
// WG tile: 128 rows x 32 cols. 8 waves: wave = m-tile, each wave owns 2 n-tiles.
// K pipeline: KB_=4 k-tiles per barrier round, double-buffered LDS (2 x 8 KB).
__global__ __launch_bounds__(256) void gemm_step(
    const __bf16* __restrict__ xfrag,
    const __bf16* __restrict__ wfrag,
    const float*  __restrict__ bias_mid,
    const float*  __restrict__ bias_pn,
    const float*  __restrict__ bias_act,
    float* __restrict__ outH,        // outputs + t*B*H : tanh(mid) = new hidden
    float* __restrict__ pushval,     // B x 1024
    float* __restrict__ actbuf)      // B x 12
{
    __shared__ __align__(32) __bf16 Bsh[2][KB_ * 2][512];   // 16 KB double buffer

    const int tid  = threadIdx.x;
    const int wave = tid >> 5;        // m-tile, and staging-chunk id
    const int lane = tid & 31;
    const int n2   = blockIdx.x;      // 0..96 (pair of n-tiles)

    // push/act columns only consume the hidden K-range [2048,4096)
    const int ktStart  = (n2 < 64) ? 0 : 64;
    const int nrounds  = (n2 < 64) ? (KT_ / KB_) : (64 / KB_);

    // this wave stages chunk (ktL = wave>>1, ntL = wave&1)
    const int cNt  = 2 * n2 + (wave & 1);
    const int cKtL = wave >> 1;

    // prologue: load round 0 chunk and park it in stage 0
    uint4 lo, hi;
    {
        const uint4* p = (const uint4*)(wfrag +
            (((size_t)cNt * KT_) + ktStart + cKtL) * 512 + (size_t)lane * 16);
        lo = p[0]; hi = p[1];
        uint4* q = (uint4*)&Bsh[0][wave][lane * 16];
        q[0] = lo; q[1] = hi;
    }

    v8f acc0 = {};
    v8f acc1 = {};
    int stage = 0;

    for (int r = 0; r < nrounds; ++r) {
        __syncthreads();                               // stage `stage` is ready

        // issue next round's global loads before computing (hide latency)
        if (r + 1 < nrounds) {
            const uint4* p = (const uint4*)(wfrag +
                (((size_t)cNt * KT_) + ktStart + (r + 1) * KB_ + cKtL) * 512 +
                (size_t)lane * 16);
            lo = p[0]; hi = p[1];
        }
        // pull round r+2's W blocks toward the WGP (global_prefetch_b8)
        if (r + 2 < nrounds) {
            __builtin_prefetch(wfrag +
                (((size_t)cNt * KT_) + ktStart + (r + 2) * KB_ + cKtL) * 512 +
                (size_t)lane * 16, 0, 1);
        }

        const int ktBase = ktStart + r * KB_;
        #pragma unroll
        for (int kl = 0; kl < KB_; ++kl) {
            const v16bf a = *(const v16bf*)(xfrag +
                ((size_t)wave * KT_ + ktBase + kl) * 512 + (size_t)lane * 16);
            const v16bf b0 = *(const v16bf*)(&Bsh[stage][kl * 2 + 0][lane * 16]);
            const v16bf b1 = *(const v16bf*)(&Bsh[stage][kl * 2 + 1][lane * 16]);
            acc0 = __builtin_amdgcn_wmma_f32_16x16x32_bf16(
                       false, a, false, b0, (short)0, acc0, false, false);
            acc1 = __builtin_amdgcn_wmma_f32_16x16x32_bf16(
                       false, a, false, b1, (short)0, acc1, false, false);
        }

        __syncthreads();                               // all waves done reading `stage`
        if (r + 1 < nrounds) {
            uint4* q = (uint4*)&Bsh[stage ^ 1][wave][lane * 16];
            q[0] = lo; q[1] = hi;
        }
        stage ^= 1;
    }

    // C/D layout: VGPR r, lane<16 -> M=r, lane>=16 -> M=r+8 ; N = lane&15
    const int rowBase = wave * 16 + ((lane >= 16) ? 8 : 0);
    #pragma unroll
    for (int ntL = 0; ntL < 2; ++ntL) {
        const int coln = n2 * 32 + ntL * 16 + (lane & 15);
        const v8f  acc = ntL ? acc1 : acc0;
        #pragma unroll
        for (int r = 0; r < 8; ++r) {
            const int row = rowBase + r;
            const float v = acc[r];
            if (coln < 2048) {
                outH[(size_t)row * H_ + coln] = tanhf(v + bias_mid[coln]);
            } else if (coln < 3072) {
                const int c = coln - 2048;
                pushval[(size_t)row * 1024 + c] = tanhf(v + bias_pn[c]);
            } else {
                const int c = coln - 3072;
                if (c < 12) actbuf[(size_t)row * 12 + c] = v + bias_act[c];
            }
        }
    }
}

// ------------- per-step stack update (in-place, carried registers) -------------
__global__ __launch_bounds__(256) void stack_update(
    float* __restrict__ stacks,
    const float* __restrict__ actbuf,
    const float* __restrict__ pushval,
    const float* __restrict__ empty_elem)
{
    int id = blockIdx.x * blockDim.x + threadIdx.x;   // B*NSTACK*E threads
    int b  = id >> 10;
    int rm = id & 1023;
    int n  = rm >> 8;
    int e  = rm & 255;

    float* p = stacks + (((size_t)(b * NSTACK_ + n) * S_) << 8) + e;
    const float ppush = actbuf[(size_t)b * 12 + n * 3 + 0];
    const float ppop  = actbuf[(size_t)b * 12 + n * 3 + 1];
    const float pnoop = actbuf[(size_t)b * 12 + n * 3 + 2];
    const float pv    = pushval[(size_t)b * 1024 + n * 256 + e];

    float old_sm1 = 0.f;
    float old_s   = p[0];
    #pragma unroll 4
    for (int s = 0; s < S_ - 1; ++s) {
        const float old_sp1 = p[(size_t)(s + 1) * E_];
        float neu;
        if (s == 0) neu = ppush * pv;
        else        neu = fmaf(ppush, old_sm1, fmaf(ppop, old_sp1, pnoop * old_s));
        p[(size_t)s * E_] = neu;
        old_sm1 = old_s;
        old_s   = old_sp1;
    }
    p[(size_t)(S_ - 1) * E_] = empty_elem[e];
}

// ---------------------------- launcher ----------------------------
extern "C" void kernel_launch(void* const* d_in, const int* in_sizes, int n_in,
                              void* d_out, int out_size, void* d_ws, size_t ws_size,
                              hipStream_t stream) {
    const int*   inputs     = (const int*)  d_in[0];   // (T,B)
    const float* hidden0    = (const float*)d_in[1];   // (B,H)
    const float* stacks0    = (const float*)d_in[2];   // (B,NSTACK,S,E)
    const float* embedding  = (const float*)d_in[3];   // (V,H)
    const float* i2h_W      = (const float*)d_in[4];
    const float* i2h_b      = (const float*)d_in[5];
    const float* h2h_W      = (const float*)d_in[6];
    const float* h2h_b      = (const float*)d_in[7];
    const float* h2a_W      = (const float*)d_in[8];   // (N,H,3)
    const float* h2a_b      = (const float*)d_in[9];   // (N,3)
    const float* h2s_W      = (const float*)d_in[10];  // (N,H,E)
    const float* h2s_b      = (const float*)d_in[11];  // (N,E)
    const float* s2h_W      = (const float*)d_in[12];  // (N,DEPTH*E,H)
    const float* s2h_b      = (const float*)d_in[13];  // (N,H)
    const float* empty_elem = (const float*)d_in[14];  // (1,E)

    float* out     = (float*)d_out;
    float* outputs = out;                                        // (T,B,H)
    float* hid_out = out + (size_t)T_ * B_ * H_;                 // (B,H)
    float* stacks  = hid_out + (size_t)B_ * H_;                  // (B,NSTACK,S,E) working+final

    // workspace carve
    __bf16* wfrag    = (__bf16*)d_ws;                            // NT_*KT_*512 bf16  (~50.9 MB)
    __bf16* xfrag    = wfrag + (size_t)NT_ * KT_ * 512;          // MT_*KT_*512 bf16  (2 MB)
    float*  bias_mid = (float*)(xfrag + (size_t)MT_ * KT_ * 512);
    float*  bias_pn  = bias_mid + H_;
    float*  bias_act = bias_pn + 1024;
    float*  pushval  = bias_act + 32;                            // B*1024
    float*  actbuf   = pushval + (size_t)B_ * 1024;              // B*12

    // seed working stacks with the input stacks
    hipMemcpyAsync(stacks, stacks0, (size_t)B_ * NSTACK_ * S_ * E_ * sizeof(float),
                   hipMemcpyDeviceToDevice, stream);

    // build bf16 swizzled weights + fused biases (once per call)
    build_wfrag<<<NT_ * KT_, 256, 0, stream>>>(wfrag, i2h_W, h2h_W, s2h_W, h2s_W, h2a_W);
    build_bias<<<(H_ + 1024 + 32 + 255) / 256, 256, 0, stream>>>(
        bias_mid, bias_pn, bias_act, i2h_b, h2h_b, s2h_b, h2s_b, h2a_b);

    for (int t = 0; t < T_; ++t) {
        const float* hprev = (t == 0) ? hidden0 : outputs + (size_t)(t - 1) * B_ * H_;
        build_xfrag<<<MT_ * KT_, 256, 0, stream>>>(xfrag, inputs, embedding, hprev, stacks, t);
        gemm_step<<<NT2_, 256, 0, stream>>>(xfrag, wfrag, bias_mid, bias_pn, bias_act,
                                            outputs + (size_t)t * B_ * H_, pushval, actbuf);
        stack_update<<<(B_ * NSTACK_ * E_) / 256, 256, 0, stream>>>(
            stacks, actbuf, pushval, empty_elem);
    }

    // final hidden = outputs[T-1]
    hipMemcpyAsync(hid_out, outputs + (size_t)(T_ - 1) * B_ * H_,
                   (size_t)B_ * H_ * sizeof(float), hipMemcpyDeviceToDevice, stream);
}
